// st_gcn_25451976196937
// MI455X (gfx1250) — matrix operand order repaired and checked by reading, not compiled, not run on hardware
//
#include <hip/hip_runtime.h>

typedef __attribute__((ext_vector_type(16))) _Float16 v16h;
typedef __attribute__((ext_vector_type(8)))  float    v8f;

#define N_  64
#define C_  64
#define T_  512
#define V_  25
#define E_  64
#define K_  9
#define F_  64
#define TV_ (T_ * V_)          // 12800
#define EPS_ 1e-5f

// ---------- fragment load helpers (contiguous b128 loads) ----------
__device__ __forceinline__ v16h load16(const _Float16* p) {
  v16h r;
  reinterpret_cast<uint4*>(&r)[0] = reinterpret_cast<const uint4*>(p)[0];
  reinterpret_cast<uint4*>(&r)[1] = reinterpret_cast<const uint4*>(p)[1];
  return r;
}
__device__ __forceinline__ v16h load8x2(const _Float16* p0, const _Float16* p1) {
  v16h r;
  reinterpret_cast<uint4*>(&r)[0] = *reinterpret_cast<const uint4*>(p0);
  reinterpret_cast<uint4*>(&r)[1] = *reinterpret_cast<const uint4*>(p1);
  return r;
}

// ---------------- K0: fold BN + biases, transpose weights to f16 ----------------
__global__ void k0_prep(const float* gcn_w, const float* conv_w, const float* gcn_b,
                        const float* bn1_g, const float* bn1_b, const float* bn1_m, const float* bn1_v,
                        const float* conv_b,
                        const float* bn2_g, const float* bn2_b, const float* bn2_m, const float* bn2_v,
                        _Float16* wt_h, _Float16* wh, float* coef) {
  int tid = threadIdx.x;
  for (int idx = tid; idx < F_ * C_; idx += blockDim.x) {       // wt[f][c] = gcn_w[c][f]
    int f = idx >> 6, c = idx & 63;
    wt_h[f * C_ + c] = (_Float16)gcn_w[c * F_ + f];
  }
  for (int idx = tid; idx < K_ * F_ * C_; idx += blockDim.x) {  // wh[k][o][i] = conv_w[o][i][k]
    int k = idx / (F_ * C_);
    int rem = idx - k * F_ * C_;
    int o = rem >> 6, i = rem & 63;
    wh[(k * F_ + o) * C_ + i] = (_Float16)conv_w[(o * C_ + i) * K_ + k];
  }
  for (int f = tid; f < F_; f += blockDim.x) {
    float a1 = bn1_g[f] * rsqrtf(bn1_v[f] + EPS_);
    coef[f]       = a1;
    coef[64 + f]  = (gcn_b[f] - bn1_m[f]) * a1 + bn1_b[f];
    float a2 = bn2_g[f] * rsqrtf(bn2_v[f] + EPS_);
    coef[128 + f] = a2;
    coef[192 + f] = (conv_b[f] - bn2_m[f]) * a2 + bn2_b[f];
  }
}

// ---------------- KN: per-sample symmetric GCN edge norms ----------------
__global__ void kn_norm(const int* edges, float* norm) {
  int n = blockIdx.x;
  if (threadIdx.x != 0) return;
  const int* es = edges + n * 2 * E_;
  float deg[V_];
  for (int v = 0; v < V_; ++v) deg[v] = 0.f;
  for (int e = 0; e < E_ + V_; ++e) {
    int d = (e < E_) ? es[E_ + e] : (e - E_);
    deg[d] += 1.f;
  }
  float dinv[V_];
  for (int v = 0; v < V_; ++v) dinv[v] = deg[v] > 0.f ? rsqrtf(deg[v]) : 0.f;
  for (int e = 0; e < E_ + V_; ++e) {
    int s = (e < E_) ? es[e]      : (e - E_);
    int d = (e < E_) ? es[E_ + e] : (e - E_);
    norm[n * 96 + e] = dinv[s] * dinv[d];
  }
}

// ---------------- K1: attention + softmax + V-mixing, per (n,t) ----------------
__global__ void k1_att(const float* X, const float* w1, const float* b1p,
                       const float* w2, const float* b2p,
                       float* Xn, _Float16* Xnh) {
  int t = blockIdx.x, n = blockIdx.y;
  int tid = threadIdx.x;
  __shared__ float xl[C_][V_];
  __shared__ float a1s[V_], a2s[V_];
  __shared__ float att[V_][V_];

  size_t base = (size_t)n * C_ * TV_ + (size_t)t * V_;   // X[n, c, t, v] stride C: TV_
  for (int idx = tid; idx < C_ * V_; idx += blockDim.x) {
    int c = idx / V_, v = idx - c * V_;
    xl[c][v] = X[base + (size_t)c * TV_ + v];
  }
  __syncthreads();

  if (tid < 2 * V_) {                                    // a1/a2 dots over channels
    int which = tid / V_, v = tid - which * V_;
    const float* w = which ? w2 : w1;
    float s = which ? b2p[0] : b1p[0];
    for (int c = 0; c < C_; ++c) s += xl[c][v] * w[c];
    if (which) a2s[v] = s; else a1s[v] = s;
  }
  __syncthreads();

  if (tid < V_) {                                        // softmax over i for column j
    int j = tid;
    float mx = -3.4e38f;
    for (int i = 0; i < V_; ++i) {
      float s = a1s[i] + a2s[j];
      s = s >= 0.f ? s : 0.01f * s;
      mx = fmaxf(mx, s);
    }
    float sum = 0.f;
    for (int i = 0; i < V_; ++i) {
      float s = a1s[i] + a2s[j];
      s = s >= 0.f ? s : 0.01f * s;
      float e = expf(s - mx);
      att[i][j] = e;
      sum += e;
    }
    float inv = 1.f / sum;
    for (int i = 0; i < V_; ++i) att[i][j] *= inv;
  }
  __syncthreads();

  size_t hbase = ((size_t)(n * T_ + t)) * V_ * C_;       // Xnh[(n,t,v), c]
  for (int idx = tid; idx < C_ * V_; idx += blockDim.x) {
    int c = idx / V_, i = idx - c * V_;
    float s = 0.f;
    for (int j = 0; j < V_; ++j) s += xl[c][j] * att[i][j];
    Xn[base + (size_t)c * TV_ + i] = s;                  // f32 residual, NCTV
    Xnh[hbase + (size_t)i * C_ + c] = (_Float16)s;       // f16, channel-innermost
  }
}

// ---------------- K2: h = Xn * gcn_w via WMMA (M=F, K=C, cols = n*t*v), f16 out ----------------
__global__ void k2_gcn_gemm(const _Float16* Xnh, const _Float16* wt_h, _Float16* h) {
  int tid = threadIdx.x;
  int wv = tid >> 5, lane = tid & 31;
  int m0 = (wv >> 1) * 16;
  int mlane = lane & 15;
  int khalf = lane >> 4;
  size_t col = ((size_t)blockIdx.x * 2 + (wv & 1)) * 16 + mlane;

  v8f acc = {};
  #pragma unroll
  for (int kg = 0; kg < C_; kg += 32) {
    const _Float16* ap = wt_h + (m0 + mlane) * C_ + kg + khalf * 16;
    v16h a = load16(ap);
    v16h b = load16(Xnh + col * C_ + kg + khalf * 16);
    acc = __builtin_amdgcn_wmma_f32_16x16x32_f16(false, a, false, b, (short)0, acc, false, false);
  }
  union { v8f v; float f[8]; } u; u.v = acc;
  union { uint4 q; _Float16 hh[8]; } s;
  #pragma unroll
  for (int r = 0; r < 8; ++r) s.hh[r] = (_Float16)u.f[r];
  *reinterpret_cast<uint4*>(h + col * F_ + m0 + khalf * 8) = s.q;  // single b128 store
}

// ---------------- K3: edge scatter-add + gcn_b + BN1 + ReLU -> f16 yh ----------------
__global__ void k3_agg(const _Float16* h, const int* edges, const float* norm,
                       const float* coef, _Float16* yh) {
  int t = blockIdx.x, n = blockIdx.y;
  int f = threadIdx.x;                                   // 64 threads, one feature lane each
  __shared__ float hl[V_][F_];
  __shared__ float acc[V_][F_];
  size_t base = ((size_t)(n * T_ + t)) * V_ * F_;
  for (int idx = f; idx < V_ * F_; idx += blockDim.x) {
    hl[idx >> 6][idx & 63]  = (float)h[base + idx];
    acc[idx >> 6][idx & 63] = 0.f;
  }
  __syncthreads();
  const int* es = edges + n * 2 * E_;
  const float* nr = norm + n * 96;
  for (int e = 0; e < E_ + V_; ++e) {                    // serial edges, exclusive f column
    int s = (e < E_) ? es[e]      : (e - E_);
    int d = (e < E_) ? es[E_ + e] : (e - E_);
    acc[d][f] += hl[s][f] * nr[e];
  }
  float a1 = coef[f], b1 = coef[64 + f];
  for (int v = 0; v < V_; ++v) {
    float x = acc[v][f] * a1 + b1;
    yh[base + v * F_ + f] = (_Float16)fmaxf(x, 0.f);
  }
}

// ---------------- K4: temporal conv (K=9) as 18 WMMAs/tile + BN2 + res + ReLU ----------------
__global__ void k4_conv(const _Float16* yh, const _Float16* wh, const float* coef,
                        const float* res, float* out) {
  int n = blockIdx.y;
  int tid = threadIdx.x;
  int wv = tid >> 5, lane = tid & 31;
  __shared__ float s_a[F_], s_b[F_];
  if (tid < F_) { s_a[tid] = coef[128 + tid]; s_b[tid] = coef[192 + tid]; }
  __syncthreads();

  int m0 = (wv >> 1) * 16;
  int mlane = lane & 15;
  int khalf = lane >> 4;
  int col = (blockIdx.x * 2 + (wv & 1)) * 16 + mlane;    // flat (t,v) column
  int t0 = col / V_;
  const _Float16* yn = yh + (size_t)n * TV_ * F_;
  size_t obase = (size_t)n * F_ * TV_ + col;

  // Prefetch the strided residual tile so its latency hides under the WMMAs.
  #pragma unroll
  for (int r = 0; r < 8; ++r) {
    int o = m0 + khalf * 8 + r;
    __builtin_prefetch(res + obase + (size_t)o * TV_, 0, 3);
  }

  v8f acc = {};
  for (int k = 0; k < K_; ++k) {
    int ts = t0 + k - 4;
    bool valid = (unsigned)ts < (unsigned)T_;
    int colS = col + (k - 4) * V_;                       // shifted flat column
    #pragma unroll
    for (int kg = 0; kg < C_; kg += 32) {
      const _Float16* ap = wh + (k * F_ + m0 + mlane) * C_ + kg + khalf * 16;
      v16h a = load16(ap);
      v16h b;
      uint4* bp = reinterpret_cast<uint4*>(&b);
      if (valid) {
        const uint4* sp = reinterpret_cast<const uint4*>(yn + (size_t)colS * F_ + kg + khalf * 16);
        bp[0] = sp[0]; bp[1] = sp[1];
      } else {
        bp[0] = make_uint4(0, 0, 0, 0); bp[1] = make_uint4(0, 0, 0, 0);
      }
      acc = __builtin_amdgcn_wmma_f32_16x16x32_f16(false, a, false, b, (short)0, acc, false, false);
    }
  }

  union { v8f v; float f[8]; } u; u.v = acc;
  #pragma unroll
  for (int r = 0; r < 8; ++r) {
    int o = m0 + khalf * 8 + r;                          // output channel (row)
    size_t idx = obase + (size_t)o * TV_;
    float val = u.f[r] * s_a[o] + s_b[o] + res[idx];
    out[idx] = fmaxf(val, 0.f);
  }
}

extern "C" void kernel_launch(void* const* d_in, const int* in_sizes, int n_in,
                              void* d_out, int out_size, void* d_ws, size_t ws_size,
                              hipStream_t stream) {
  (void)in_sizes; (void)n_in; (void)out_size; (void)ws_size;
  const float* X      = (const float*)d_in[0];
  const int*   edges  = (const int*)  d_in[1];
  const float* w_att1 = (const float*)d_in[2];
  const float* b_att1 = (const float*)d_in[3];
  const float* w_att2 = (const float*)d_in[4];
  const float* b_att2 = (const float*)d_in[5];
  const float* gcn_w  = (const float*)d_in[6];
  const float* gcn_b  = (const float*)d_in[7];
  const float* bn1_g  = (const float*)d_in[8];
  const float* bn1_b  = (const float*)d_in[9];
  const float* bn1_m  = (const float*)d_in[10];
  const float* bn1_v  = (const float*)d_in[11];
  const float* conv_w = (const float*)d_in[12];
  const float* conv_b = (const float*)d_in[13];
  const float* bn2_g  = (const float*)d_in[14];
  const float* bn2_b  = (const float*)d_in[15];
  const float* bn2_m  = (const float*)d_in[16];
  const float* bn2_v  = (const float*)d_in[17];

  char* w = (char*)d_ws;
  const size_t szXn  = (size_t)N_ * C_ * TV_ * 4;        // 209,715,200  (f32 residual)
  const size_t szXnh = (size_t)N_ * TV_ * C_ * 2;        // 104,857,600  (f16 GEMM input / yh)
  const size_t szH   = (size_t)N_ * TV_ * F_ * 2;        // 104,857,600  (f16 GEMM output)
  float*    Xn   = (float*)w;
  _Float16* Xnh  = (_Float16*)(w + szXn);
  _Float16* h    = (_Float16*)(w + szXn + szXnh);
  _Float16* yh   = Xnh;                                  // alias: Xnh dead after K2
  size_t off = szXn + szXnh + szH;
  _Float16* wt_h = (_Float16*)(w + off);  off += (size_t)F_ * C_ * 2;
  _Float16* wh   = (_Float16*)(w + off);  off += (size_t)K_ * F_ * C_ * 2;
  float*    norm = (float*)(w + off);     off += (size_t)N_ * 96 * 4;
  float*    coef = (float*)(w + off);

  k0_prep<<<1, 256, 0, stream>>>(gcn_w, conv_w, gcn_b,
                                 bn1_g, bn1_b, bn1_m, bn1_v,
                                 conv_b, bn2_g, bn2_b, bn2_m, bn2_v,
                                 wt_h, wh, coef);
  kn_norm<<<N_, 32, 0, stream>>>(edges, norm);
  k1_att<<<dim3(T_, N_), 128, 0, stream>>>(X, w_att1, b_att1, w_att2, b_att2, Xn, Xnh);
  k2_gcn_gemm<<<(N_ * TV_) / 32, 256, 0, stream>>>(Xnh, wt_h, h);
  k3_agg<<<dim3(T_, N_), 64, 0, stream>>>(h, edges, norm, coef, yh);
  k4_conv<<<dim3(TV_ / 32, N_), 256, 0, stream>>>(yh, wh, coef, Xn, (float*)d_out);
}
